// ScaledDotProductAttention_35519379538643
// MI455X (gfx1250) — matrix operand order repaired and checked by reading, not compiled, not run on hardware
//
#include <hip/hip_runtime.h>

typedef __attribute__((ext_vector_type(16))) __bf16 v16bf;
typedef __attribute__((ext_vector_type(8)))  float  v8f;
typedef __attribute__((ext_vector_type(4)))  float  v4f;

namespace {
constexpr int kB = 4, kH = 16, kS = 2048, kD = 128;
constexpr int kRowsPerWave  = 16;
constexpr int kWaves        = 8;
constexpr int kRowsPerBlock = kRowsPerWave * kWaves;   // 128
constexpr int kQBlocks      = kS / kRowsPerBlock;      // 16
// 1/sqrt(128) * log2(e): scores land in log2 domain -> softmax uses native v_exp_f32 (exp2)
constexpr float kScaleQ = 0.08838834764831845f * 1.4426950408889634f;
}

union Frag16 { uint4 q[2]; v16bf v; };   // 32B: one 16x32 bf16 WMMA fragment

__device__ __forceinline__ float fast_exp2(float x) {
#if __has_builtin(__builtin_amdgcn_exp2f)
  return __builtin_amdgcn_exp2f(x);      // raw v_exp_f32
#else
  return exp2f(x);
#endif
}

__device__ __forceinline__ void nt_store4(float4 v, float* p) {
  union { float4 f; v4f n; } u;
  u.f = v;
  __builtin_nontemporal_store(u.n, (v4f*)p);   // global_store_b128, TH=NT
}

__device__ __forceinline__ unsigned pk_bf16(float lo, float hi) {
#if __has_builtin(__builtin_amdgcn_cvt_pk_bf16_f32)
  auto r = __builtin_amdgcn_cvt_pk_bf16_f32(lo, hi);   // single v_cvt_pk_bf16_f32
  union { decltype(r) v; unsigned u; } x;
  x.v = r;
  return x.u;
#else
  union { __bf16 h[2]; unsigned u; } x;
  x.h[0] = (__bf16)lo; x.h[1] = (__bf16)hi;
  return x.u;
#endif
}
__device__ __forceinline__ uint4 pk4(float4 a, float4 b, float s) {
  uint4 u;
  u.x = pk_bf16(a.x * s, a.y * s);
  u.y = pk_bf16(a.z * s, a.w * s);
  u.z = pk_bf16(b.x * s, b.y * s);
  u.w = pk_bf16(b.z * s, b.w * s);
  return u;
}

__global__ __launch_bounds__(256) void sdpa_wmma_bf16_kernel(
    const float* __restrict__ Q, const float* __restrict__ K,
    const float* __restrict__ V, float* __restrict__ Out,
    float* __restrict__ Attn) {
  // Cooperative staging buffers (bf16), padded rows: 16B-aligned, bank-spread.
  __shared__ unsigned short Klds[32][136];    // [key][feature]      8704 B
  __shared__ unsigned short Vlds[128][40];    // [feature][key] (T) 10240 B
  __shared__ float ldsP[kWaves][16][36];      // per-wave P tile    18432 B

  const int lane = threadIdx.x & 31;
  const int wave = threadIdx.x >> 5;
  const int bh   = blockIdx.x >> 4;               // b*H + h
  const int qblk = blockIdx.x & (kQBlocks - 1);

  const float* Qb = Q + (size_t)bh * kS * kD;
  const float* Kb = K + (size_t)bh * kS * kD;
  const float* Vb = V + (size_t)bh * kS * kD;
  float* Ob = Out  + (size_t)bh * kS * kD;
  float* Ab = Attn + (size_t)bh * kS * kS;

  const int col = lane & 15;
  const int hiA = (lane & 16) ? 8  : 0;   // 16-bit A-frag K sub-offset
  const int hiB = (lane & 16) ? 16 : 0;   // 16-bit B-frag K offset
  const int hiC = (lane & 16) ? 8  : 0;   // C/D row offset

  const int q0   = qblk * kRowsPerBlock + wave * kRowsPerWave;
  const int qrow = q0 + col;

  // ---- Q strip 16x128 -> 4 bf16 A-fragments, pre-scaled by log2(e)/sqrt(dk) ----
  Frag16 aq[4];
  #pragma unroll
  for (int c = 0; c < 4; ++c) {
    const float* qp = Qb + (size_t)qrow * kD + c * 32;
    const float4* qa = (const float4*)(qp + hiA);
    const float4* qb = (const float4*)(qp + 16 + hiA);
    aq[c].q[0] = pk4(qa[0], qa[1], kScaleQ);
    aq[c].q[1] = pk4(qb[0], qb[1], kScaleQ);
  }

  // ---- cooperative stagers (all 256 threads) ----
  auto stageK = [&](int kc) {   // 32 keys x 128 feats, row-major bf16
    const int k  = threadIdx.x & 31;
    const int fg = threadIdx.x >> 5;    // 0..7 -> 16 feats each
    const float4* src = (const float4*)(Kb + (size_t)(kc * 32 + k) * kD + fg * 16);
    float4 x0 = src[0], x1 = src[1], x2 = src[2], x3 = src[3];
    uint4* dst = (uint4*)&Klds[k][fg * 16];
    dst[0] = pk4(x0, x1, 1.0f);
    dst[1] = pk4(x2, x3, 1.0f);
  };
  auto stageV = [&](int kc) {   // transpose to [feature][key], bf16 pairs along key
    const int kp = threadIdx.x & 15;    // key pair: k0 = 2*kp
    const int fg = threadIdx.x >> 4;    // 0..15 -> 8 feats each
    const float* r0 = Vb + (size_t)(kc * 32 + 2 * kp) * kD + fg * 8;
    const float* r1 = r0 + kD;
    float4 a0 = ((const float4*)r0)[0], a1 = ((const float4*)r0)[1];
    float4 b0 = ((const float4*)r1)[0], b1 = ((const float4*)r1)[1];
    unsigned w[8] = { pk_bf16(a0.x, b0.x), pk_bf16(a0.y, b0.y),
                      pk_bf16(a0.z, b0.z), pk_bf16(a0.w, b0.w),
                      pk_bf16(a1.x, b1.x), pk_bf16(a1.y, b1.y),
                      pk_bf16(a1.z, b1.z), pk_bf16(a1.w, b1.w) };
    #pragma unroll
    for (int j = 0; j < 8; ++j)
      *(unsigned*)&Vlds[fg * 8 + j][2 * kp] = w[j];
  };

  // ---- fragment readers: 2x ds_load_b128 each ----
  auto kfrag = [&](int sub, int c) -> v16bf {
    Frag16 f;
    f.q[0] = *(const uint4*)&Klds[sub * 16 + col][c * 32 + hiA];
    f.q[1] = *(const uint4*)&Klds[sub * 16 + col][c * 32 + 16 + hiA];
    return f.v;
  };
  auto vfrag = [&](int t) -> v16bf {
    Frag16 f;
    f.q[0] = *(const uint4*)&Vlds[t * 16 + col][hiA];
    f.q[1] = *(const uint4*)&Vlds[t * 16 + col][16 + hiA];
    return f.v;
  };
  auto score_tile = [&](int sub) -> v8f {
    v8f acc = {};
    #pragma unroll
    for (int c = 0; c < 4; ++c) {
      v16bf bk = kfrag(sub, c);
      acc = __builtin_amdgcn_wmma_f32_16x16x32_bf16(
          false, aq[c].v, false, bk, (short)0, acc, false, false);
    }
    return acc;
  };

  // ---- Pass 1: per-lane online softmax stats (no cross-lane ops in hot loop) ----
  float m[8], l[8];
  #pragma unroll
  for (int r = 0; r < 8; ++r) { m[r] = -3.0e38f; l[r] = 0.0f; }

  for (int kc = 0; kc < kS / 32; ++kc) {
    stageK(kc);
    __syncthreads();
    #pragma unroll
    for (int sub = 0; sub < 2; ++sub) {
      v8f s = score_tile(sub);
      #pragma unroll
      for (int r = 0; r < 8; ++r) {
        float x  = s[r];
        float mn = fmaxf(m[r], x);
        l[r] = l[r] * fast_exp2(m[r] - mn) + fast_exp2(x - mn);
        m[r] = mn;
      }
    }
    __syncthreads();
  }
  // One-time 16-lane merge (xor 1,2,4,8 stays inside each half)
  float invl[8];
  #pragma unroll
  for (int r = 0; r < 8; ++r) {
    #pragma unroll
    for (int mask = 1; mask <= 8; mask <<= 1) {
      float m2 = __shfl_xor(m[r], mask, 32);
      float l2 = __shfl_xor(l[r], mask, 32);
      float mn = fmaxf(m[r], m2);
      l[r] = l[r] * fast_exp2(m[r] - mn) + l2 * fast_exp2(m2 - mn);
      m[r] = mn;
    }
    invl[r] = 1.0f / l[r];
  }

  // ---- Pass 2: recompute scores, write P (vectorized NT), accumulate O = P x V ----
  v8f o[8];
  #pragma unroll
  for (int t = 0; t < 8; ++t) o[t] = (v8f){};

  for (int kc = 0; kc < kS / 32; ++kc) {
    stageK(kc);
    stageV(kc);
    __syncthreads();

    #pragma unroll
    for (int sub = 0; sub < 2; ++sub) {
      v8f s = score_tile(sub);
      #pragma unroll
      for (int r = 0; r < 8; ++r)
        ldsP[wave][hiC + r][sub * 16 + col] = fast_exp2(s[r] - m[r]) * invl[r];
    }
    __syncthreads();

    // Attention writes: each lane owns one 64B line -> 4 NT b128 stores
    {
      const int r = lane & 15;
      const int h = lane >> 4;
      const float* srcp = &ldsP[wave][r][h * 16];
      float4 p0 = ((const float4*)srcp)[0];
      float4 p1 = ((const float4*)srcp)[1];
      float4 p2 = ((const float4*)srcp)[2];
      float4 p3 = ((const float4*)srcp)[3];
      float* dst = Ab + (size_t)(q0 + r) * kS + kc * 32 + h * 16;
      nt_store4(p0, dst);
      nt_store4(p1, dst + 4);
      nt_store4(p2, dst + 8);
      nt_store4(p3, dst + 12);
    }

    // P chunk in A-fragment layout (bf16)
    Frag16 ap;
    {
      const float* pr = &ldsP[wave][col][0];
      const float4* pa = (const float4*)(pr + hiA);
      const float4* pb = (const float4*)(pr + 16 + hiA);
      ap.q[0] = pk4(pa[0], pa[1], 1.0f);
      ap.q[1] = pk4(pb[0], pb[1], 1.0f);
    }

    // O[16x128] += P(16x32) x V(32x128)
    #pragma unroll
    for (int t = 0; t < 8; ++t) {
      v16bf bv = vfrag(t);
      o[t] = __builtin_amdgcn_wmma_f32_16x16x32_bf16(
          false, ap.v, false, bv, (short)0, o[t], false, false);
    }
    __syncthreads();
  }

  // ---- Write O (written exactly once; non-temporal) ----
  #pragma unroll
  for (int t = 0; t < 8; ++t) {
    #pragma unroll
    for (int r = 0; r < 8; ++r) {
      __builtin_nontemporal_store(
          o[t][r], Ob + (size_t)(q0 + hiC + r) * kD + t * 16 + col);
    }
  }
}

extern "C" void kernel_launch(void* const* d_in, const int* in_sizes, int n_in,
                              void* d_out, int out_size, void* d_ws, size_t ws_size,
                              hipStream_t stream) {
  const float* Q = (const float*)d_in[0];
  const float* K = (const float*)d_in[1];
  const float* V = (const float*)d_in[2];
  float* Out  = (float*)d_out;
  float* Attn = Out + (size_t)kB * kH * kS * kD;  // (output, attention) concatenated

  dim3 grid(kB * kH * kQBlocks);   // 1024 workgroups
  dim3 block(256);                 // 8 wave32 waves
  sdpa_wmma_bf16_kernel<<<grid, block, 0, stream>>>(Q, K, V, Out, Attn);
}
